// LinearAttention_57561151701689
// MI455X (gfx1250) — compile-verified, matrix operand order checked by reading, not run on hardware
//
#include <hip/hip_runtime.h>
#include <hip/hip_bf16.h>

// ---------------------------------------------------------------------------
// Types for CDNA5 WMMA (wave32): v_wmma_f32_16x16x32_bf16 + TDM descriptors
// ---------------------------------------------------------------------------
typedef __bf16 bf16_t;
typedef __attribute__((ext_vector_type(16))) __bf16 v16bf;
typedef __attribute__((ext_vector_type(8)))  __bf16 v8bf;
typedef __attribute__((ext_vector_type(8)))  float  v8f;
typedef __attribute__((ext_vector_type(4)))  unsigned int v4u;
typedef __attribute__((ext_vector_type(8)))  int v8i;
typedef __attribute__((ext_vector_type(4)))  int v4i;

#define NHEAD 8
#define DIM_HEAD 64
#define BATCH 16
#define CIN 256
#define NPIX 4096           // 64*64
#define OQKV 1536           // 3*8*64
#define EOUT 512            // 8*64
#define SCALE 0.125f        // 64^-0.5

__device__ __forceinline__ bf16_t f2bf(float f) {
    unsigned u = __builtin_bit_cast(unsigned, f);
    unsigned r = (u + 0x7FFFu + ((u >> 16) & 1u)) >> 16;
    unsigned short h = (unsigned short)r;
    return __builtin_bit_cast(bf16_t, h);
}

__device__ __forceinline__ v8f zero8() {
    v8f z;
#pragma unroll
    for (int i = 0; i < 8; ++i) z[i] = 0.0f;
    return z;
}

// ---------------------------------------------------------------------------
// TDM descriptor construction (CDNA5 ISA ch.8: D# group0 / group1, 2-D tile)
// ---------------------------------------------------------------------------
__device__ __forceinline__ unsigned lds_off32(const void* p) {
    // generic pointers to LDS carry the workgroup-relative offset in [31:0]
    return (unsigned)(unsigned long long)(uintptr_t)p;
}

__device__ __forceinline__ v4u tdm_g0(unsigned lds_addr, const void* gptr) {
    const unsigned long long ga = (unsigned long long)(uintptr_t)gptr;
    v4u g;
    g[0] = 1u;                                            // count=1, user mode
    g[1] = lds_addr;                                      // [63:32] lds_addr
    g[2] = (unsigned)(ga & 0xFFFFFFFFu);                  // global_addr[31:0]
    g[3] = (unsigned)((ga >> 32) & 0x01FFFFFFu) | 0x80000000u; // [56:32]|type=2
    return g;
}

// 2-D tile of 2-byte elements: tile_d0 contiguous elems, tile_d1 rows,
// row stride stride0 (elements).  Written linearly into LDS.
__device__ __forceinline__ v8i tdm_g1_2d(unsigned tensor_d0, unsigned tensor_d1,
                                         unsigned tile_d0, unsigned tile_d1,
                                         unsigned stride0) {
    v8i g;
    g[0] = (int)(1u << 16);                               // data_size = 2 bytes
    g[1] = (int)((tensor_d0 & 0xFFFFu) << 16);            // dim0[15:0] -> [31:16]
    g[2] = (int)(((tensor_d0 >> 16) & 0xFFFFu) |
                 ((tensor_d1 & 0xFFFFu) << 16));          // dim0[31:16] | dim1[15:0]
    g[3] = (int)(((tensor_d1 >> 16) & 0xFFFFu) |
                 ((tile_d0 & 0xFFFFu) << 16));            // dim1[31:16] | tile_d0
    g[4] = (int)(tile_d1 & 0xFFFFu);                      // tile_d1, tile_d2=0
    g[5] = (int)stride0;                                  // dim0_stride[31:0]
    g[6] = 0;                                             // stride hi / dim1_stride
    g[7] = 0;
    return g;
}

__device__ __forceinline__ void tdm_load_2d(void* lds_dst, const void* gsrc,
                                            unsigned tensor_d0, unsigned tensor_d1,
                                            unsigned tile_d0, unsigned tile_d1,
                                            unsigned stride0) {
    const v4i z4 = {0, 0, 0, 0};
    const v8i z8 = {0, 0, 0, 0, 0, 0, 0, 0};
    // 6-arg toolchain form: (g0, g1, g2, g3, g4, cpol)
    __builtin_amdgcn_tensor_load_to_lds(
        tdm_g0(lds_off32(lds_dst), gsrc),
        tdm_g1_2d(tensor_d0, tensor_d1, tile_d0, tile_d1, stride0),
        z4, z4, z8, 0);
}

// ---------------------------------------------------------------------------
// WMMA fragment helpers (ISA 7.12.2 layouts)
// ---------------------------------------------------------------------------
__device__ __forceinline__ v16bf load_frag(const bf16_t* base, int stride) {
    const int lane = threadIdx.x & 31;
    const int r    = lane & 15;
    const int koff = (lane >> 4) << 3;      // 0 or 8
    const bf16_t* p = base + r * stride + koff;
    v8bf lo = *(const v8bf*)(p);
    v8bf hi = *(const v8bf*)(p + 16);
    return __builtin_shufflevector(lo, hi, 0, 1, 2, 3, 4, 5, 6, 7,
                                           8, 9, 10, 11, 12, 13, 14, 15);
}

__device__ __forceinline__ v8f wmma_bf16(v16bf a, v16bf b, v8f c) {
    return __builtin_amdgcn_wmma_f32_16x16x32_bf16(false, a, false, b,
                                                   (short)0, c, false, false);
}

// ---------------------------------------------------------------------------
// Kernel 1: convert weights to bf16
// ---------------------------------------------------------------------------
__global__ __launch_bounds__(256) void k_convert_weights(
    const float* __restrict__ wqkv, const float* __restrict__ wout,
    bf16_t* __restrict__ wq_bf, bf16_t* __restrict__ wo_bf) {
    const int stride = gridDim.x * blockDim.x;
    const int tid = blockIdx.x * blockDim.x + threadIdx.x;
    for (int i = tid; i < OQKV * CIN; i += stride) wq_bf[i] = f2bf(wqkv[i]);
    for (int i = tid; i < CIN * EOUT; i += stride) wo_bf[i] = f2bf(wout[i]);
}

// ---------------------------------------------------------------------------
// Kernel 2: qkv[b,o,n] = sum_c wqkv[o,c] * x[b,c,n]    (M=1536,K=256,N=4096)
// A tile (bf16 weights) streamed by TDM; B tile converted fp32->bf16 manually.
// ---------------------------------------------------------------------------
__global__ __launch_bounds__(256) void k_qkv_gemm(
    const float* __restrict__ x, const bf16_t* __restrict__ wq,
    float* __restrict__ qkv) {
    __shared__ __align__(16) bf16_t lds_a[128 * 32];   // [o][c] K-contig
    __shared__ __align__(16) bf16_t lds_b[128 * 32];   // [n][c] K-contig

    const int b  = blockIdx.z;
    const int m0 = blockIdx.y * 128;
    const int n0 = blockIdx.x * 128;
    const float* xb = x   + (size_t)b * CIN  * NPIX;
    float*       qb = qkv + (size_t)b * OQKV * NPIX;

    const int tid  = threadIdx.x;
    const int wid  = tid >> 5;
    const int lane = tid & 31;
    const int wm   = (wid >> 1) * 32;
    const int wn   = (wid & 1) * 64;

    v8f acc[2][4];
#pragma unroll
    for (int i = 0; i < 2; ++i)
#pragma unroll
        for (int j = 0; j < 4; ++j) acc[i][j] = zero8();

    for (int k0 = 0; k0 < CIN; k0 += 32) {
        if (wid == 0) {                                  // TDM: A tile 128x32
            tdm_load_2d(lds_a, &wq[(size_t)m0 * CIN + k0],
                        CIN, OQKV - m0, 32, 128, CIN);
        }
        // B tile: x[c][n] fp32 -> bf16, stored transposed [n][c]
        for (int i = tid; i < 32 * 32; i += 256) {
            const int c = i >> 5, n4 = (i & 31) * 4;
            const float4 f = *(const float4*)&xb[(size_t)(k0 + c) * NPIX + n0 + n4];
            lds_b[(n4 + 0) * 32 + c] = f2bf(f.x);
            lds_b[(n4 + 1) * 32 + c] = f2bf(f.y);
            lds_b[(n4 + 2) * 32 + c] = f2bf(f.z);
            lds_b[(n4 + 3) * 32 + c] = f2bf(f.w);
        }
        if (wid == 0) __builtin_amdgcn_s_wait_tensorcnt(0);
        __syncthreads();
#pragma unroll
        for (int i = 0; i < 2; ++i) {
            const v16bf a = load_frag(&lds_a[(wm + i * 16) * 32], 32);
#pragma unroll
            for (int j = 0; j < 4; ++j) {
                const v16bf bb = load_frag(&lds_b[(wn + j * 16) * 32], 32);
                acc[i][j] = wmma_bf16(a, bb, acc[i][j]);
            }
        }
        __syncthreads();
    }
    const int l = lane & 15, mb = (lane >> 4) * 8;
#pragma unroll
    for (int i = 0; i < 2; ++i)
#pragma unroll
        for (int j = 0; j < 4; ++j)
#pragma unroll
            for (int e = 0; e < 8; ++e)
                qb[(size_t)(m0 + wm + i * 16 + mb + e) * NPIX +
                   n0 + wn + j * 16 + l] = acc[i][j][e];
}

// ---------------------------------------------------------------------------
// Kernel 3: fused linear attention per (b,h); writes att TRANSPOSED:
//   att_t[b][n][e_full]  (e_full = h*64+e), bf16, 16B-packed stores
// ---------------------------------------------------------------------------
__global__ __launch_bounds__(256) void k_attn_fused(
    const float* __restrict__ qkv, bf16_t* __restrict__ att_t) {
    __shared__ float rowmax[64], rowsum[64];
    __shared__ float colmax[128], colsum[128];
    __shared__ __align__(16) bf16_t lds_p[64 * 128];    // softmaxed k  [d][n]
    __shared__ __align__(16) bf16_t lds_v[64 * 128];    // v            [e][n]
    __shared__ __align__(16) float  lds_qraw[64 * 128]; // raw q        [d][n]
    __shared__ __align__(16) bf16_t lds_qs[128 * 64];   // softmaxed q  [n][d]
    __shared__ __align__(16) bf16_t lds_phi[64 * 64];   // phi^T        [e][d]

    const int b = blockIdx.x >> 3;
    const int h = blockIdx.x & 7;
    const float* qptr = qkv + ((size_t)b * OQKV + (size_t)h * 64) * NPIX;
    const float* kptr = qptr + (size_t)512  * NPIX;
    const float* vptr = qptr + (size_t)1024 * NPIX;
    bf16_t* attb = att_t + (size_t)b * NPIX * EOUT;     // [n][e_full]

    const int tid = threadIdx.x, wid = tid >> 5, lane = tid & 31;

    // ---- Phase A: online row softmax stats for k ----
    for (int r = wid * 8; r < wid * 8 + 8; ++r) {
        const float* krow = kptr + (size_t)r * NPIX;
        float m = -INFINITY, s = 0.0f;
        for (int n = lane; n < NPIX; n += 32) {
            const float v = krow[n];
            const float nm = fmaxf(m, v);
            s = s * __expf(m - nm) + __expf(v - nm);
            m = nm;
        }
#pragma unroll
        for (int off = 16; off > 0; off >>= 1) {
            const float om = __shfl_xor(m, off, 32);
            const float os = __shfl_xor(s, off, 32);
            const float nm = fmaxf(m, om);
            s = s * __expf(m - nm) + os * __expf(om - nm);
            m = nm;
        }
        if (lane == 0) { rowmax[r] = m; rowsum[r] = s; }
    }
    __syncthreads();

    // ---- Phase B: phi[d,e] accumulation streamed over n ----
    v8f phi_acc[2] = { zero8(), zero8() };
    for (int nc = 0; nc < NPIX; nc += 128) {
        for (int i = tid; i < 64 * 32; i += 256) {
            const int d = i >> 5, n4 = (i & 31) * 4;
            const float4 f = *(const float4*)&kptr[(size_t)d * NPIX + nc + n4];
            const float mx = rowmax[d], inv = 1.0f / rowsum[d];
            lds_p[d * 128 + n4 + 0] = f2bf(__expf(f.x - mx) * inv);
            lds_p[d * 128 + n4 + 1] = f2bf(__expf(f.y - mx) * inv);
            lds_p[d * 128 + n4 + 2] = f2bf(__expf(f.z - mx) * inv);
            lds_p[d * 128 + n4 + 3] = f2bf(__expf(f.w - mx) * inv);
        }
        for (int i = tid; i < 64 * 32; i += 256) {
            const int e = i >> 5, n4 = (i & 31) * 4;
            const float4 f = *(const float4*)&vptr[(size_t)e * NPIX + nc + n4];
            lds_v[e * 128 + n4 + 0] = f2bf(f.x);
            lds_v[e * 128 + n4 + 1] = f2bf(f.y);
            lds_v[e * 128 + n4 + 2] = f2bf(f.z);
            lds_v[e * 128 + n4 + 3] = f2bf(f.w);
        }
        __syncthreads();
#pragma unroll
        for (int kk = 0; kk < 128; kk += 32) {
            const v16bf a = load_frag(&lds_p[((wid >> 1) * 16) * 128 + kk], 128);
#pragma unroll
            for (int j = 0; j < 2; ++j) {
                const v16bf bb =
                    load_frag(&lds_v[((wid & 1) * 32 + j * 16) * 128 + kk], 128);
                phi_acc[j] = wmma_bf16(a, bb, phi_acc[j]);
            }
        }
        __syncthreads();
    }

    // ---- Phase C: phi^T -> LDS ----
    {
        const int l = lane & 15, mb = (lane >> 4) * 8;
#pragma unroll
        for (int j = 0; j < 2; ++j)
#pragma unroll
            for (int i2 = 0; i2 < 8; ++i2) {
                const int d = (wid >> 1) * 16 + mb + i2;
                const int e = (wid & 1) * 32 + j * 16 + l;
                lds_phi[e * 64 + d] = f2bf(phi_acc[j][i2]);
            }
    }
    __syncthreads();

    // ---- Phase D: att[e,n] = sum_d phi[d,e] q_s[d,n], stored transposed ----
    for (int nc = 0; nc < NPIX; nc += 128) {
        for (int i = tid; i < 64 * 32; i += 256) {
            const int d = i >> 5, n4 = (i & 31) * 4;
            *(float4*)&lds_qraw[d * 128 + n4] =
                *(const float4*)&qptr[(size_t)d * NPIX + nc + n4];
        }
        __syncthreads();
        if (tid < 128) {
            float m = -INFINITY;
            for (int d = 0; d < 64; ++d) m = fmaxf(m, lds_qraw[d * 128 + tid]);
            float s = 0.0f;
            for (int d = 0; d < 64; ++d) s += __expf(lds_qraw[d * 128 + tid] - m);
            colmax[tid] = m; colsum[tid] = s;
        }
        __syncthreads();
        for (int i = tid; i < 128 * 16; i += 256) {
            const int n = i >> 4, d0 = (i & 15) * 4;
            const float mx = colmax[n], inv = SCALE / colsum[n];
#pragma unroll
            for (int u = 0; u < 4; ++u)
                lds_qs[n * 64 + d0 + u] =
                    f2bf(__expf(lds_qraw[(d0 + u) * 128 + n] - mx) * inv);
        }
        __syncthreads();
        v8f oacc[4] = { zero8(), zero8(), zero8(), zero8() };
#pragma unroll
        for (int kk = 0; kk < 64; kk += 32) {
            const v16bf a = load_frag(&lds_phi[((wid >> 1) * 16) * 64 + kk], 64);
#pragma unroll
            for (int j = 0; j < 4; ++j) {
                const v16bf bb =
                    load_frag(&lds_qs[((wid & 1) * 64 + j * 16) * 64 + kk], 64);
                oacc[j] = wmma_bf16(a, bb, oacc[j]);
            }
        }
        // store transposed: att_t[n][h*64 + e], 8 contiguous e per lane -> b128
        const int l = lane & 15, mb = (lane >> 4) * 8;
        const int e0 = h * 64 + (wid >> 1) * 16 + mb;
#pragma unroll
        for (int j = 0; j < 4; ++j) {
            const int n = nc + (wid & 1) * 64 + j * 16 + l;
            v8bf pack;
#pragma unroll
            for (int i2 = 0; i2 < 8; ++i2) pack[i2] = f2bf(oacc[j][i2]);
            *(v8bf*)&attb[(size_t)n * EOUT + e0] = pack;
        }
        __syncthreads();
    }
}

// ---------------------------------------------------------------------------
// Kernel 4: out[b,c,n] = sum_e wout[c,e] * att_t[b,n,e]  (M=256,K=512,N=4096)
// Both tiles are K-contiguous 2-D bf16 tiles -> TDM, double-buffered.
// ---------------------------------------------------------------------------
__global__ __launch_bounds__(256) void k_out_gemm(
    const bf16_t* __restrict__ att_t, const bf16_t* __restrict__ wo,
    float* __restrict__ out) {
    __shared__ __align__(16) bf16_t lds_a[2][128 * 32];  // [c][e] K-contig
    __shared__ __align__(16) bf16_t lds_b[2][128 * 32];  // [n][e] K-contig

    const int b  = blockIdx.z;
    const int m0 = blockIdx.y * 128;
    const int n0 = blockIdx.x * 128;
    const bf16_t* attb = att_t + (size_t)b * NPIX * EOUT;
    float*        ob   = out   + (size_t)b * CIN  * NPIX;

    const int tid = threadIdx.x, wid = tid >> 5, lane = tid & 31;
    const int wm = (wid >> 1) * 32, wn = (wid & 1) * 64;

    v8f acc[2][4];
#pragma unroll
    for (int i = 0; i < 2; ++i)
#pragma unroll
        for (int j = 0; j < 4; ++j) acc[i][j] = zero8();

    const int KT = EOUT / 32;                            // 16 K-tiles
    if (wid == 0) {                                      // prologue: tile 0
        tdm_load_2d(lds_a[0], &wo[(size_t)m0 * EOUT], EOUT, CIN - m0,
                    32, 128, EOUT);
        tdm_load_2d(lds_b[0], &attb[(size_t)n0 * EOUT], EOUT, NPIX - n0,
                    32, 128, EOUT);
        __builtin_amdgcn_s_wait_tensorcnt(0);
    }
    __syncthreads();

    for (int kt = 0; kt < KT; ++kt) {
        const int cur = kt & 1;
        if (wid == 0 && kt + 1 < KT) {                   // prefetch next tile
            const int kn = (kt + 1) * 32;
            tdm_load_2d(lds_a[cur ^ 1], &wo[(size_t)m0 * EOUT + kn],
                        EOUT, CIN - m0, 32, 128, EOUT);
            tdm_load_2d(lds_b[cur ^ 1], &attb[(size_t)n0 * EOUT + kn],
                        EOUT, NPIX - n0, 32, 128, EOUT);
        }
#pragma unroll
        for (int i = 0; i < 2; ++i) {
            const v16bf a = load_frag(&lds_a[cur][(wm + i * 16) * 32], 32);
#pragma unroll
            for (int j = 0; j < 4; ++j) {
                const v16bf bb = load_frag(&lds_b[cur][(wn + j * 16) * 32], 32);
                acc[i][j] = wmma_bf16(a, bb, acc[i][j]);
            }
        }
        if (wid == 0 && kt + 1 < KT) __builtin_amdgcn_s_wait_tensorcnt(0);
        __syncthreads();
    }
    const int l = lane & 15, mb = (lane >> 4) * 8;
#pragma unroll
    for (int i = 0; i < 2; ++i)
#pragma unroll
        for (int j = 0; j < 4; ++j)
#pragma unroll
            for (int e = 0; e < 8; ++e)
                ob[(size_t)(m0 + wm + i * 16 + mb + e) * NPIX +
                   n0 + wn + j * 16 + l] = acc[i][j][e];
}

// ---------------------------------------------------------------------------
// Launch
// ---------------------------------------------------------------------------
extern "C" void kernel_launch(void* const* d_in, const int* in_sizes, int n_in,
                              void* d_out, int out_size, void* d_ws, size_t ws_size,
                              hipStream_t stream) {
    const float* x    = (const float*)d_in[0];
    const float* wqkv = (const float*)d_in[1];
    const float* wout = (const float*)d_in[2];
    float* out = (float*)d_out;

    char* ws = (char*)d_ws;
    bf16_t* wq_bf = (bf16_t*)ws;                                   // 768 KB
    bf16_t* wo_bf = (bf16_t*)(ws + (size_t)OQKV * CIN * 2);        // 256 KB
    float*  qkv   = (float*)(ws + (1u << 20));                     // 384 MB
    bf16_t* att_t = (bf16_t*)(ws + (1u << 20) +
                              (size_t)BATCH * OQKV * NPIX * 4);    // 64 MB

    k_convert_weights<<<512, 256, 0, stream>>>(wqkv, wout, wq_bf, wo_bf);
    k_qkv_gemm<<<dim3(NPIX / 128, OQKV / 128, BATCH), 256, 0, stream>>>(x, wq_bf, qkv);
    k_attn_fused<<<dim3(BATCH * NHEAD), 256, 0, stream>>>(qkv, att_t);
    k_out_gemm<<<dim3(NPIX / 128, CIN / 128, BATCH), 256, 0, stream>>>(att_t, wo_bf, out);
}
